// Self_Attention_48206712930842
// MI455X (gfx1250) — matrix operand order repaired and checked by reading, not compiled, not run on hardware
//
#include <hip/hip_runtime.h>
#include <hip/hip_bf16.h>

// ---------------------------------------------------------------------------
// CDNA5 (gfx1250) self-attention, bf16 WMMA pipeline, fp32 accumulate.
// ---------------------------------------------------------------------------

typedef __attribute__((ext_vector_type(16))) __bf16 v16bf;
typedef __attribute__((ext_vector_type(8)))  float  v8f;

struct alignas(16) U4 { unsigned int x, y, z, w; };
union FragU { U4 u[2]; v16bf v; };

#define DIM   1024
#define SEQ   2048
#define BATCH 4

// fp32 -> bf16, round-to-nearest-even
__device__ inline unsigned short f2bf(float f) {
  unsigned int u = __float_as_uint(f);
  u += 0x7FFFu + ((u >> 16) & 1u);
  return (unsigned short)(u >> 16);
}
__device__ inline unsigned int pack2(float a, float b) {
  return (unsigned int)f2bf(a) | ((unsigned int)f2bf(b) << 16);
}

// A-matrix fragment (16x32 bf16, MxK). Layout: lanes 0-15 hold row M=lane,
// K = k0+0..15 contiguous; lanes 16-31 hold K = k0+16..31.
__device__ inline v16bf load_frag_a(const unsigned short* base, int ld,
                                    int row0, int k0, int lane) {
  const int r   = lane & 15;
  const int sel = lane >> 4;
  const unsigned short* p = base + (size_t)(row0 + r) * ld + k0 + sel * 16;
  FragU f;
  f.u[0] = *reinterpret_cast<const U4*>(p);        // K = +0..7  (of this half)
  f.u[1] = *reinterpret_cast<const U4*>(p + 8);    // K = +8..15
  return f.v;
}

// B-matrix fragment (32x16 bf16, KxN), operand stored row-major [N][K] in mem
// (i.e. we compute A * B^T). Layout used here (same k order as A): lanes 0-15 hold column N=lane with
// K = k0+0..15 contiguous; lanes 16-31 hold K = k0+16..31.
__device__ inline v16bf load_frag_b(const unsigned short* base, int ld,
                                    int n0, int k0, int lane) {
  const int r   = lane & 15;
  const int sel = lane >> 4;
  const unsigned short* p = base + (size_t)(n0 + r) * ld + k0 + sel * 16;
  FragU f;
  f.u[0] = *reinterpret_cast<const U4*>(p);        // K = +0..7
  f.u[1] = *reinterpret_cast<const U4*>(p + 8);    // K = +8..15
  return f.v;
}

__device__ inline v8f wmma_bf16(v16bf a, v16bf b, v8f c) {
  // (neg_a, A, neg_b, B, c_mod, C, reuse_a, reuse_b)
  return __builtin_amdgcn_wmma_f32_16x16x32_bf16(false, a, false, b,
                                                 (short)0, c, false, false);
}

// ---------------------------------------------------------------------------
// Kernel 1: fp32 -> bf16 elementwise conversion
// ---------------------------------------------------------------------------
__global__ void cvt_f32_bf16(const float* __restrict__ in,
                             unsigned short* __restrict__ out, int n) {
  int i = blockIdx.x * blockDim.x + threadIdx.x;
  if (i < n) out[i] = f2bf(in[i]);
}

// ---------------------------------------------------------------------------
// Kernel 2: y = x * W^T + b   (x:[M,1024] bf16, W:[1024,1024] bf16 row-major
// over K, bias fp32).  transposeOut==0 -> row-major bf16 [M,1024] (Q,K);
// transposeOut==1 -> per-batch transposed bf16 Vt[b][v][n] (V).
// grid = (M/16, 1024/512), block = 256 (8 waves); each wave owns 16x64 of D.
// ---------------------------------------------------------------------------
__global__ __launch_bounds__(256) void qkv_gemm(
    const unsigned short* __restrict__ A,
    const unsigned short* __restrict__ W,
    const float* __restrict__ bias,
    unsigned short* __restrict__ outRM,
    unsigned short* __restrict__ outT,
    int transposeOut) {
  const int lane = threadIdx.x & 31;
  const int wave = threadIdx.x >> 5;
  const int m0 = blockIdx.x * 16;
  const int n0 = blockIdx.y * 512 + wave * 64;
  const int r = lane & 15, sel = lane >> 4;

  const v8f z = {0.f,0.f,0.f,0.f,0.f,0.f,0.f,0.f};
  v8f acc[4];
#pragma unroll
  for (int j = 0; j < 4; ++j) acc[j] = z;

  const unsigned short* Abase = A + (size_t)m0 * DIM;
  for (int kk = 0; kk < 32; ++kk) {
    const int k0 = kk * 32;
    v16bf af = load_frag_a(Abase, DIM, 0, k0, lane);
#pragma unroll
    for (int j = 0; j < 4; ++j) {
      v16bf bf = load_frag_b(W, DIM, n0 + j * 16, k0, lane);
      acc[j] = wmma_bf16(af, bf, acc[j]);
    }
  }

#pragma unroll
  for (int j = 0; j < 4; ++j) {
    const int n = n0 + j * 16 + r;          // D: lane -> N
    const float bv = bias[n];
    v8f a = acc[j];
    if (!transposeOut) {
#pragma unroll
      for (int t = 0; t < 8; ++t) {         // D: VGPR t -> M = t + 8*sel
        const int m = m0 + t + 8 * sel;
        outRM[(size_t)m * DIM + n] = f2bf(a[t] + bv);
      }
    } else {
      const int b   = m0 >> 11;             // m0 / 2048 (tiles never straddle)
      const int col = (m0 & (SEQ - 1)) + 8 * sel;
      unsigned short* dst = outT + ((size_t)b * DIM + n) * SEQ + col;
      U4 pk;
      pk.x = pack2(a[0] + bv, a[1] + bv);
      pk.y = pack2(a[2] + bv, a[3] + bv);
      pk.z = pack2(a[4] + bv, a[5] + bv);
      pk.w = pack2(a[6] + bv, a[7] + bv);
      *reinterpret_cast<U4*>(dst) = pk;     // 8 consecutive m -> b128 store
    }
  }
}

// ---------------------------------------------------------------------------
// Kernel 3: fused scores + softmax*norm + P*V for a 16-query tile.
// grid = (SEQ/16, BATCH), block = 256 (8 waves), dynamic LDS = 16*2048*4 B.
// Phase 1: S(16x2048 fp32) into LDS via WMMA (wave w -> keys [w*256,w*256+256))
// Phase 2: row softmax (wave32 shfl reductions), write bf16 P in place
// Phase 3: out = P * V via WMMA with A-fragments from LDS, B from Vt (global)
// ---------------------------------------------------------------------------
__global__ __launch_bounds__(256) void attn_fused(
    const unsigned short* __restrict__ Qh,   // [B*SEQ, DIM] bf16
    const unsigned short* __restrict__ Kh,   // [B*SEQ, DIM] bf16
    const unsigned short* __restrict__ Vt,   // [B][DIM][SEQ] bf16
    float* __restrict__ out) {               // [B*SEQ, DIM] fp32
  extern __shared__ char smem[];
  float*          Sf = reinterpret_cast<float*>(smem);          // [16][2048]
  unsigned short* Pb = reinterpret_cast<unsigned short*>(smem); // row m @ m*4096

  const int lane = threadIdx.x & 31;
  const int wave = threadIdx.x >> 5;
  const int r = lane & 15, sel = lane >> 4;
  const int q0 = blockIdx.x * 16;
  const int b  = blockIdx.y;

  const unsigned short* Qbase = Qh + ((size_t)b * SEQ + q0) * DIM;
  const unsigned short* Kbase = Kh + (size_t)b * SEQ * DIM;
  const unsigned short* Vbase = Vt + (size_t)b * DIM * SEQ;

  const v8f z = {0.f,0.f,0.f,0.f,0.f,0.f,0.f,0.f};

  // ---- Phase 1: S = Q K^T -------------------------------------------------
  for (int g = 0; g < 4; ++g) {
    const int nbase = wave * 256 + g * 64;
    v8f acc[4];
#pragma unroll
    for (int j = 0; j < 4; ++j) acc[j] = z;
    for (int kk = 0; kk < 32; ++kk) {
      const int k0 = kk * 32;
      v16bf af = load_frag_a(Qbase, DIM, 0, k0, lane);
#pragma unroll
      for (int j = 0; j < 4; ++j) {
        v16bf bf = load_frag_b(Kbase, DIM, nbase + j * 16, k0, lane);
        acc[j] = wmma_bf16(af, bf, acc[j]);
      }
    }
#pragma unroll
    for (int j = 0; j < 4; ++j)
#pragma unroll
      for (int t = 0; t < 8; ++t)
        Sf[(t + 8 * sel) * SEQ + nbase + j * 16 + r] = acc[j][t];
  }
  __syncthreads();

  // ---- Phase 2: softmax, then * 1/sqrt(1024) (reference order) -----------
  const float norm_fact = 0.03125f;
  for (int rr = 0; rr < 2; ++rr) {
    const int m = wave * 2 + rr;
    float vals[64];
    float mx = -3.0e38f;
#pragma unroll
    for (int i = 0; i < 64; ++i) {
      float v = Sf[m * SEQ + i * 32 + lane];
      vals[i] = v;
      mx = fmaxf(mx, v);
    }
    for (int off = 16; off > 0; off >>= 1)
      mx = fmaxf(mx, __shfl_xor(mx, off, 32));
    float s = 0.f;
#pragma unroll
    for (int i = 0; i < 64; ++i) { vals[i] = __expf(vals[i] - mx); s += vals[i]; }
    for (int off = 16; off > 0; off >>= 1)
      s += __shfl_xor(s, off, 32);
    const float scale = norm_fact / s;
#pragma unroll
    for (int i = 0; i < 64; ++i)           // entire row already in registers:
      Pb[m * 4096 + i * 32 + lane] = f2bf(vals[i] * scale);  // in-place safe
  }
  __syncthreads();

  // ---- Phase 3: out = P * V  (B operand from transposed Vt) ---------------
  for (int g = 0; g < 2; ++g) {
    const int vb = wave * 128 + g * 64;
    v8f acc[4];
#pragma unroll
    for (int j = 0; j < 4; ++j) acc[j] = z;
    for (int nc = 0; nc < 64; ++nc) {
      const int k0 = nc * 32;
      v16bf af = load_frag_a(Pb, 4096, 0, k0, lane);   // ds_load_b128 x2
#pragma unroll
      for (int j = 0; j < 4; ++j) {
        v16bf bf = load_frag_b(Vbase, SEQ, vb + j * 16, k0, lane);
        acc[j] = wmma_bf16(af, bf, acc[j]);
      }
    }
#pragma unroll
    for (int j = 0; j < 4; ++j)
#pragma unroll
      for (int t = 0; t < 8; ++t)
        out[((size_t)b * SEQ + q0 + t + 8 * sel) * DIM + vb + j * 16 + r] =
            acc[j][t];
  }
}

// ---------------------------------------------------------------------------
extern "C" void kernel_launch(void* const* d_in, const int* in_sizes, int n_in,
                              void* d_out, int out_size, void* d_ws,
                              size_t ws_size, hipStream_t stream) {
  const float* x  = (const float*)d_in[0];
  const float* Wq = (const float*)d_in[1];
  const float* bq = (const float*)d_in[2];
  const float* Wk = (const float*)d_in[3];
  const float* bk = (const float*)d_in[4];
  const float* Wv = (const float*)d_in[5];
  const float* bv = (const float*)d_in[6];
  float* out = (float*)d_out;

  // Workspace layout (bf16 stages), 70 MB total
  char* ws = (char*)d_ws;
  unsigned short* xh  = (unsigned short*)(ws);                       // 16 MB
  unsigned short* wqh = (unsigned short*)(ws + (size_t)16 * 1048576); // 2 MB
  unsigned short* wkh = (unsigned short*)(ws + (size_t)18 * 1048576); // 2 MB
  unsigned short* wvh = (unsigned short*)(ws + (size_t)20 * 1048576); // 2 MB
  unsigned short* qh  = (unsigned short*)(ws + (size_t)22 * 1048576); // 16 MB
  unsigned short* kh  = (unsigned short*)(ws + (size_t)38 * 1048576); // 16 MB
  unsigned short* vth = (unsigned short*)(ws + (size_t)54 * 1048576); // 16 MB

  const int n_x = BATCH * SEQ * DIM;   // 8388608
  const int n_w = DIM * DIM;           // 1048576

  cvt_f32_bf16<<<n_x / 256, 256, 0, stream>>>(x,  xh,  n_x);
  cvt_f32_bf16<<<n_w / 256, 256, 0, stream>>>(Wq, wqh, n_w);
  cvt_f32_bf16<<<n_w / 256, 256, 0, stream>>>(Wk, wkh, n_w);
  cvt_f32_bf16<<<n_w / 256, 256, 0, stream>>>(Wv, wvh, n_w);

  const int M = BATCH * SEQ;                      // 8192
  dim3 ggrid(M / 16, DIM / 512);                  // (512, 2)
  qkv_gemm<<<ggrid, 256, 0, stream>>>(xh, wqh, bq, qh, nullptr, 0);
  qkv_gemm<<<ggrid, 256, 0, stream>>>(xh, wkh, bk, kh, nullptr, 0);
  qkv_gemm<<<ggrid, 256, 0, stream>>>(xh, wvh, bv, nullptr, vth, 1);

  const int smem_bytes = 16 * SEQ * 4;            // 128 KB (WGP has 320 KB)
  (void)hipFuncSetAttribute((const void*)attn_fused,
                            hipFuncAttributeMaxDynamicSharedMemorySize,
                            smem_bytes);
  attn_fused<<<dim3(SEQ / 16, BATCH), 256, smem_bytes, stream>>>(qh, kh, vth,
                                                                 out);
}
